// GNNModel_43954695307621
// MI455X (gfx1250) — matrix operand order repaired and checked by reading, not compiled, not run on hardware
//
#include <hip/hip_runtime.h>
#include <hip/hip_bf16.h>

// ---------------------------------------------------------------------------
// GNN (NNConv x2 + edge head) for MI455X / gfx1250.
// Dominant cost: [E,4096]@[4096,4096] edge-MLP GEMM (~1.1 TFLOP) -> bf16 WMMA,
// fused with the per-edge [1,64]x[64,64] contraction so the E x 4096
// intermediate never touches HBM. Uses 320KB WGP LDS for the A-tile.
// ---------------------------------------------------------------------------

#define N_NODES 10000
#define N_EDGES 32768
#define IN_F    6
#define HID     64
#define EA_F    4

typedef __bf16 bf16;
typedef __bf16 v16bf __attribute__((ext_vector_type(16)));
typedef float  v8f   __attribute__((ext_vector_type(8)));
typedef unsigned int v4u __attribute__((ext_vector_type(4)));

union FragBF { v16bf v; v4u u[2]; };

__device__ __forceinline__ v8f wmma_bf16(v16bf a, v16bf b, v8f c) {
  // D = A(16x32 bf16) * B(32x16 bf16) + C(16x16 f32)
  return __builtin_amdgcn_wmma_f32_16x16x32_bf16(false, a, false, b,
                                                 (short)0, c, false, false);
}

// ---------------------------------------------------------------------------
// Workspace zeroing (atomic accumulators must start at 0 every launch).
__global__ void k_zero(float* s1, float* s2, float* cnt) {
  int stride = gridDim.x * blockDim.x;
  int i0 = blockIdx.x * blockDim.x + threadIdx.x;
  for (int i = i0; i < N_NODES * HID; i += stride) { s1[i] = 0.f; s2[i] = 0.f; }
  for (int i = i0; i < N_NODES; i += stride) cnt[i] = 0.f;
}

// Transpose + convert fp32 weight [K,C] -> bf16 Bt[C,K] (this code's B fragments load
// 16 contiguous K per lane for a fixed column).
__global__ void k_convT(const float* __restrict__ w, bf16* __restrict__ bt,
                        int K, int C) {
  long long total = (long long)K * C;
  long long stride = (long long)gridDim.x * blockDim.x;
  for (long long idx = (long long)blockIdx.x * blockDim.x + threadIdx.x;
       idx < total; idx += stride) {
    int c = (int)(idx / K);
    int k = (int)(idx % K);
    bt[idx] = (bf16)w[(long long)k * C + c];
  }
}

// In-degree counts (shared by both scatter-means).
__global__ void k_count(const int* __restrict__ ei, float* cnt) {
  int e = blockIdx.x * blockDim.x + threadIdx.x;
  if (e < N_EDGES) atomicAdd(&cnt[ei[N_EDGES + e]], 1.0f);
}

// ---------------------------------------------------------------------------
// Layer 1: hid1 = relu(ea@e1w1+b1) [E,384]; W = hid1@e1w2+b2 -> [E,6,64];
// msg = x[src] . W fused into the GEMM epilogue; scatter-add into s1.
// WG: 128 threads = 4 waves, 64 edges; each wave owns 16 edges, full K=384.
__launch_bounds__(128)
__global__ void k_layer1(const float* __restrict__ x, const int* __restrict__ ei,
                         const float* __restrict__ ea,
                         const float* __restrict__ e1w1, const float* __restrict__ e1b1,
                         const bf16* __restrict__ bt1, const float* __restrict__ e1b2,
                         float* __restrict__ s1) {
  const int KD = IN_F * HID;        // 384
  const int RS = KD + 8;            // bf16 row stride: pad to break bank conflicts
  extern __shared__ __align__(16) char smem[];
  bf16*  sHid = (bf16*)smem;                                  // [64][RS]
  float* sX   = (float*)(smem + (size_t)64 * RS * sizeof(bf16)); // [64][IN_F]
  int*   sDst = (int*)(sX + 64 * IN_F);                       // [64]

  const int tid = threadIdx.x;
  const int ebase = blockIdx.x * 64;

  for (int idx = tid; idx < 64 * IN_F; idx += 128) {
    int eL = idx / IN_F, f = idx % IN_F;
    sX[idx] = x[(size_t)ei[ebase + eL] * IN_F + f];
  }
  for (int eL = tid; eL < 64; eL += 128) sDst[eL] = ei[N_EDGES + ebase + eL];

  for (int idx = tid; idx < 64 * KD; idx += 128) {
    int eL = idx / KD, j = idx % KD;
    const float* eap = ea + (size_t)(ebase + eL) * EA_F;
    float v = e1b1[j];
#pragma unroll
    for (int t = 0; t < EA_F; ++t) v += eap[t] * e1w1[t * KD + j];
    sHid[eL * RS + j] = (bf16)fmaxf(v, 0.f);
  }
  __syncthreads();

  const int lane = tid & 31;
  const int w    = tid >> 5;
  const int r16  = lane & 15;
  const int hi   = (lane >= 16) ? 1 : 0;
  const bf16* aRow = sHid + (size_t)(w * 16 + r16) * RS;

  v8f macc[4] = {};                         // msg accumulator: 16 edges x 64
#pragma unroll 1
  for (int i = 0; i < IN_F; ++i) {
    float xs[8];
#pragma unroll
    for (int j = 0; j < 8; ++j) xs[j] = sX[(w * 16 + j + hi * 8) * IN_F + i];
#pragma unroll
    for (int ot = 0; ot < 4; ++ot) {
      v8f wa = {};
#pragma unroll
      for (int kt = 0; kt < KD / 32; ++kt) {
        const int k0 = kt * 32;
        FragBF fa, fb;
        const bf16* ap = aRow + k0 + hi * 16;
        fa.u[0] = *(const v4u*)(ap);
        fa.u[1] = *(const v4u*)(ap + 8);
        const int col = i * HID + ot * 16 + r16;
        const bf16* bp = bt1 + (size_t)col * KD + k0 + hi * 16;
        fb.u[0] = *(const v4u*)(bp);
        fb.u[1] = *(const v4u*)(bp + 8);
        wa = wmma_bf16(fa.v, fb.v, wa);
      }
      const float b2 = e1b2[i * HID + ot * 16 + r16];
#pragma unroll
      for (int j = 0; j < 8; ++j) macc[ot][j] += xs[j] * (wa[j] + b2);
    }
  }
#pragma unroll
  for (int ot = 0; ot < 4; ++ot)
#pragma unroll
    for (int j = 0; j < 8; ++j) {
      const int eL = w * 16 + j + hi * 8;
      atomicAdd(&s1[(size_t)sDst[eL] * HID + ot * 16 + r16], macc[ot][j]);
    }
}

// Node update 1: h1 = relu(s1/max(cnt,1) + x@root1 + b1), stored bf16.
__global__ void k_node1(const float* __restrict__ x, const float* __restrict__ s1,
                        const float* __restrict__ cnt, const float* __restrict__ root1,
                        const float* __restrict__ bias1, bf16* __restrict__ h1) {
  int idx = blockIdx.x * blockDim.x + threadIdx.x;
  if (idx >= N_NODES * HID) return;
  int n = idx >> 6, o = idx & 63;
  float v = s1[idx] / fmaxf(cnt[n], 1.0f) + bias1[o];
#pragma unroll
  for (int t = 0; t < IN_F; ++t) v += x[(size_t)n * IN_F + t] * root1[t * HID + o];
  h1[idx] = (bf16)fmaxf(v, 0.f);
}

// ---------------------------------------------------------------------------
// Layer 2 (the 1.1 TFLOP kernel), fully fused:
//   hid2 = relu(ea@e2w1+b1) [32 edges x 4096] bf16 in LDS (257KB, CDNA5-only)
//   msg[e,o] = sum_i hs[e,i] * (hid2[e,:] . e2w2[:, i*64+o] + b2[i*64+o])
// WG: 256 threads = 8 waves = (2 edge halves) x (4 K-quarters of 1024).
// Each wave: 64 i x 32 kt x 4 otile = 8192 v_wmma_f32_16x16x32_bf16.
__launch_bounds__(256)
__global__ void k_layer2(const bf16* __restrict__ h1, const int* __restrict__ ei,
                         const float* __restrict__ ea,
                         const float* __restrict__ e2w1, const float* __restrict__ e2b1,
                         const bf16* __restrict__ bt2, const float* __restrict__ e2b2,
                         float* __restrict__ s2) {
  const int KD = HID * HID;         // 4096
  const int RS = KD + 8;            // 4104 bf16 row stride (bank-conflict pad)
  extern __shared__ __align__(16) char smem[];
  bf16*  sHid = (bf16*)smem;                                     // [32][RS]
  float* sHS  = (float*)(smem + (size_t)32 * RS * sizeof(bf16)); // [32][64]
  float* sMsg = sHS + 32 * HID;                                  // [32][64]
  float* sEA  = sMsg + 32 * HID;                                 // [32][4]
  int*   sDst = (int*)(sEA + 32 * EA_F);                         // [32]

  const int tid = threadIdx.x;
  const int ebase = blockIdx.x * 32;

  for (int idx = tid; idx < 32 * EA_F; idx += 256)
    sEA[idx] = ea[(size_t)ebase * EA_F + idx];
  for (int idx = tid; idx < 32 * HID; idx += 256) {
    int eL = idx >> 6, i = idx & 63;
    sHS[idx]  = (float)h1[(size_t)ei[ebase + eL] * HID + i];
    sMsg[idx] = 0.f;
  }
  for (int eL = tid; eL < 32; eL += 256) sDst[eL] = ei[N_EDGES + ebase + eL];
  __syncthreads();

  for (int idx = tid; idx < 32 * KD; idx += 256) {
    int eL = idx >> 12, j = idx & (KD - 1);
    const float* eap = sEA + eL * EA_F;
    float v = e2b1[j];
#pragma unroll
    for (int t = 0; t < EA_F; ++t) v += eap[t] * e2w1[t * KD + j];
    sHid[(size_t)eL * RS + j] = (bf16)fmaxf(v, 0.f);
  }
  __syncthreads();

  const int lane = tid & 31;
  const int w    = tid >> 5;
  const int eh   = w & 1;           // which 16-edge half
  const int kq   = w >> 1;          // K quarter: [kq*1024, kq*1024+1024)
  const int r16  = lane & 15;
  const int hi   = (lane >= 16) ? 1 : 0;
  const bf16* aRow = sHid + (size_t)(eh * 16 + r16) * RS + kq * 1024;

  v8f macc[4] = {};
#pragma unroll 1
  for (int i = 0; i < HID; ++i) {
    if (i + 1 < HID) {  // pull next i-block of B toward L0/L2
      const bf16* pf = bt2 + (size_t)((i + 1) * HID + r16) * KD + kq * 1024 + hi * 16;
      __builtin_prefetch(pf, 0, 1);
    }
    v8f wa[4] = {};
#pragma unroll 1
    for (int kt = 0; kt < 32; ++kt) {
      const int k0 = kt * 32;
      FragBF fa;
      const bf16* ap = aRow + k0 + hi * 16;
      fa.u[0] = *(const v4u*)(ap);
      fa.u[1] = *(const v4u*)(ap + 8);
#pragma unroll
      for (int ot = 0; ot < 4; ++ot) {
        const int col = i * HID + ot * 16 + r16;
        const bf16* bp = bt2 + (size_t)col * KD + kq * 1024 + k0 + hi * 16;
        FragBF fb;
        fb.u[0] = *(const v4u*)(bp);
        fb.u[1] = *(const v4u*)(bp + 8);
        wa[ot] = wmma_bf16(fa.v, fb.v, wa[ot]);
      }
    }
    // fold h_src[e,i] into msg (linear in GEMM output -> valid per K-quarter)
#pragma unroll
    for (int j = 0; j < 8; ++j) {
      const float hsv = sHS[(eh * 16 + j + hi * 8) * HID + i];
#pragma unroll
      for (int ot = 0; ot < 4; ++ot) macc[ot][j] += hsv * wa[ot][j];
    }
  }
  if (kq == 0) {  // e2b2 contribution: sum_i hs[e,i]*b2[i*64+o], once per edge
#pragma unroll
    for (int ot = 0; ot < 4; ++ot) {
      const int o = ot * 16 + r16;
#pragma unroll
      for (int j = 0; j < 8; ++j) {
        const int eL = eh * 16 + j + hi * 8;
        float acc = 0.f;
        for (int i = 0; i < HID; ++i) acc += sHS[eL * HID + i] * e2b2[i * HID + o];
        macc[ot][j] += acc;
      }
    }
  }
#pragma unroll
  for (int ot = 0; ot < 4; ++ot)
#pragma unroll
    for (int j = 0; j < 8; ++j) {
      const int eL = eh * 16 + j + hi * 8;
      atomicAdd(&sMsg[eL * HID + ot * 16 + r16], macc[ot][j]);  // ds_add_f32
    }
  __syncthreads();
  for (int idx = tid; idx < 32 * HID; idx += 256) {
    int eL = idx >> 6, o = idx & 63;
    atomicAdd(&s2[(size_t)sDst[eL] * HID + o], sMsg[idx]);
  }
}

// Node update 2: h2 = relu(s2/max(cnt,1) + h1@root2 + b2), fp32.
__global__ void k_node2(const bf16* __restrict__ h1, const float* __restrict__ s2,
                        const float* __restrict__ cnt, const float* __restrict__ root2,
                        const float* __restrict__ bias2, float* __restrict__ h2) {
  int idx = blockIdx.x * blockDim.x + threadIdx.x;
  if (idx >= N_NODES * HID) return;
  int n = idx >> 6, o = idx & 63;
  float v = s2[idx] / fmaxf(cnt[n], 1.0f) + bias2[o];
  for (int k = 0; k < HID; ++k)
    v += (float)h1[(size_t)n * HID + k] * root2[k * HID + o];
  h2[idx] = fmaxf(v, 0.f);
}

// Edge head: sigmoid(hs.hd), softmax([hs,hd,ea]@fcw+fcb), plus src/dst copies.
__global__ void k_head(const float* __restrict__ h2, const int* __restrict__ ei,
                       const float* __restrict__ ea, const float* __restrict__ fcw,
                       const float* __restrict__ fcb, float* __restrict__ out,
                       int out_size) {
  int e = blockIdx.x * blockDim.x + threadIdx.x;
  if (e >= N_EDGES) return;
  const int s = ei[e], d = ei[N_EDGES + e];
  const float* hs = h2 + (size_t)s * HID;
  const float* hd = h2 + (size_t)d * HID;
  float dot = 0.f;
  float lg[3] = { fcb[0], fcb[1], fcb[2] };
  for (int k = 0; k < HID; ++k) {
    const float a = hs[k], b = hd[k];
    dot += a * b;
#pragma unroll
    for (int c = 0; c < 3; ++c)
      lg[c] += a * fcw[k * 3 + c] + b * fcw[(HID + k) * 3 + c];
  }
#pragma unroll
  for (int t = 0; t < EA_F; ++t) {
    const float av = ea[(size_t)e * EA_F + t];
#pragma unroll
    for (int c = 0; c < 3; ++c) lg[c] += av * fcw[(2 * HID + t) * 3 + c];
  }
  out[e] = 1.f / (1.f + __expf(-dot));
  const float m  = fmaxf(lg[0], fmaxf(lg[1], lg[2]));
  const float e0 = __expf(lg[0] - m), e1 = __expf(lg[1] - m), e2v = __expf(lg[2] - m);
  const float inv = 1.f / (e0 + e1 + e2v);
  float* tp = out + N_EDGES + (size_t)e * 3;
  tp[0] = e0 * inv; tp[1] = e1 * inv; tp[2] = e2v * inv;
  if (out_size >= 6 * N_EDGES) {
    out[4 * N_EDGES + e] = (float)s;
    out[5 * N_EDGES + e] = (float)d;
  }
}

// ---------------------------------------------------------------------------
extern "C" void kernel_launch(void* const* d_in, const int* in_sizes, int n_in,
                              void* d_out, int out_size, void* d_ws, size_t ws_size,
                              hipStream_t stream) {
  (void)in_sizes; (void)n_in; (void)ws_size;
  const float* x     = (const float*)d_in[0];
  const int*   ei    = (const int*)d_in[1];
  const float* ea    = (const float*)d_in[2];
  const float* e1w1  = (const float*)d_in[3];
  const float* e1b1  = (const float*)d_in[4];
  const float* e1w2  = (const float*)d_in[5];
  const float* e1b2  = (const float*)d_in[6];
  const float* root1 = (const float*)d_in[7];
  const float* bias1 = (const float*)d_in[8];
  const float* e2w1  = (const float*)d_in[9];
  const float* e2b1  = (const float*)d_in[10];
  const float* e2w2  = (const float*)d_in[11];
  const float* e2b2  = (const float*)d_in[12];
  const float* root2 = (const float*)d_in[13];
  const float* bias2 = (const float*)d_in[14];
  const float* fcw   = (const float*)d_in[15];
  const float* fcb   = (const float*)d_in[16];
  float* out = (float*)d_out;

  char* p = (char*)d_ws;
  bf16*  Bt2 = (bf16*)p;  p += (size_t)4096 * 4096 * sizeof(bf16);   // 32 MB (fits L2)
  bf16*  Bt1 = (bf16*)p;  p += (size_t)384 * 384 * sizeof(bf16);
  bf16*  h1  = (bf16*)p;  p += (size_t)N_NODES * HID * sizeof(bf16);
  float* s1  = (float*)p; p += (size_t)N_NODES * HID * sizeof(float);
  float* s2  = (float*)p; p += (size_t)N_NODES * HID * sizeof(float);
  float* h2  = (float*)p; p += (size_t)N_NODES * HID * sizeof(float);
  float* cnt = (float*)p; p += (size_t)N_NODES * sizeof(float);

  const size_t lds1 = (size_t)64 * (384 + 8) * sizeof(bf16)
                    + (size_t)64 * IN_F * sizeof(float) + 64 * sizeof(int);
  const size_t lds2 = (size_t)32 * (4096 + 8) * sizeof(bf16)
                    + (size_t)32 * HID * sizeof(float) * 2
                    + (size_t)32 * EA_F * sizeof(float) + 32 * sizeof(int);
  // CDNA5 allows up to 320KB LDS per workgroup; opt in past the 64KB default.
  (void)hipFuncSetAttribute((const void*)k_layer2,
                            hipFuncAttributeMaxDynamicSharedMemorySize, (int)lds2);

  k_zero <<<512, 256, 0, stream>>>(s1, s2, cnt);
  k_convT<<<4096, 256, 0, stream>>>(e2w2, Bt2, 4096, 4096);
  k_convT<<<144, 256, 0, stream>>>(e1w2, Bt1, 384, 384);
  k_count<<<(N_EDGES + 255) / 256, 256, 0, stream>>>(ei, cnt);
  k_layer1<<<N_EDGES / 64, 128, lds1, stream>>>(x, ei, ea, e1w1, e1b1, Bt1, e1b2, s1);
  k_node1<<<(N_NODES * HID + 255) / 256, 256, 0, stream>>>(x, s1, cnt, root1, bias1, h1);
  k_layer2<<<N_EDGES / 32, 256, lds2, stream>>>(h1, ei, ea, e2w1, e2b1, Bt2, e2b2, s2);
  k_node2<<<(N_NODES * HID + 255) / 256, 256, 0, stream>>>(h1, s2, cnt, root2, bias2, h2);
  k_head<<<(N_EDGES + 255) / 256, 256, 0, stream>>>(h2, ei, ea, fcw, fcb, out, out_size);
}
